// Metric_Loss_7516192768928
// MI455X (gfx1250) — matrix operand order repaired and checked by reading, not compiled, not run on hardware
//
#include <hip/hip_runtime.h>

typedef __attribute__((ext_vector_type(16))) _Float16 v16h;
typedef __attribute__((ext_vector_type(8)))  _Float16 v8h;
typedef __attribute__((ext_vector_type(8)))  float    v8f;

#define N_EMB   4096
#define D_EMB   1024
#define P_PAIRS 2048
#define LOSS_MARGIN 0.5f

#define KS 64     // K halves staged per iteration
#define RS 72     // padded LDS row stride in halves (144B: 16B-aligned, conflict-free)

// ---------------------------------------------------------------------------
// Kernel 1: fp32 -> fp16 conversion + interleaved text/shape matrix.
// embeddings[2p] = text[2p], embeddings[2p+1] = shape[p].
// ---------------------------------------------------------------------------
__global__ void build_half_kernel(const float* __restrict__ text,
                                  const float* __restrict__ shape,
                                  _Float16* __restrict__ Xtt,
                                  _Float16* __restrict__ Xst) {
    int i   = blockIdx.x * blockDim.x + threadIdx.x;   // over N_EMB*D_EMB
    int row = i >> 10;                                  // D_EMB == 1024
    int k   = i & (D_EMB - 1);
    float t = text[i];
    Xtt[i] = (_Float16)t;
    float v = (row & 1) ? shape[((row >> 1) << 10) + k] : t;
    Xst[i] = (_Float16)v;
}

// ---------------------------------------------------------------------------
// Async global -> LDS staging helpers (CDNA5 ASYNCcnt path)
// ---------------------------------------------------------------------------
__device__ __forceinline__ void async_ld16(unsigned ldsByteOff,
                                           const _Float16* gptr) {
    asm volatile("global_load_async_to_lds_b128 %0, %1, off"
                 :: "v"(ldsByteOff), "v"(gptr)
                 : "memory");
}

__device__ __forceinline__ void wait_async0() {
    asm volatile("s_wait_asynccnt 0x0" ::: "memory");
}

// ---------------------------------------------------------------------------
// Kernel 2: D = X * X^T via v_wmma_f32_16x16x32_f16 with double-buffered
// async LDS staging, fused with exp(margin + D), 2x2 diag-block exclusion,
// per-pair masked row sums (S) and positive-pair similarity (Dij).
//
// Block: 256 threads = 8 waves; block tile 128x128; wave tile 32x64
// (2 M-frags x 4 N-frags of 16x16); K staged in 64-wide panels.
// ---------------------------------------------------------------------------
__global__ __launch_bounds__(256)
void gram_loss_kernel(const _Float16* __restrict__ X,
                      float* __restrict__ S,
                      float* __restrict__ Dij) {
    __shared__ _Float16 shA[2][128 * RS];
    __shared__ _Float16 shB[2][128 * RS];
    __shared__ float    rowsum[128][2];

    const int tid   = threadIdx.x;
    const int lane  = tid & 31;
    const int w     = tid >> 5;
    const int waveM = w >> 1;        // 0..3
    const int waveN = w & 1;         // 0..1
    const int half  = lane >> 4;     // lane group 0/1
    const int nl    = lane & 15;

    const int blockRow = blockIdx.y * 128;
    const int blockCol = blockIdx.x * 128;

    const unsigned baseA[2] = { (unsigned)(size_t)&shA[0][0],
                                (unsigned)(size_t)&shA[1][0] };
    const unsigned baseB[2] = { (unsigned)(size_t)&shB[0][0],
                                (unsigned)(size_t)&shB[1][0] };

    v8f acc[2][4];
#pragma unroll
    for (int mi = 0; mi < 2; ++mi)
#pragma unroll
        for (int ni = 0; ni < 4; ++ni)
            acc[mi][ni] = (v8f){0.f, 0.f, 0.f, 0.f, 0.f, 0.f, 0.f, 0.f};

    // A 16-bit 16x32 layout: lanes 0-15 hold K {0..7,16..23},
    // lanes 16-31 hold K {8..15,24..31}  -> two contiguous 8-half runs.
    const int aKoff = half * 8;
    // B 16-bit 32x16 layout: lanes 0-15 K {0..7,16..23}, lanes 16-31 K {8..15,24..31}.
    const int bKoff = half * 8;

    // ---- stage one 64-wide K panel (A rows + B rows) asynchronously ------
    auto stage = [&](int buf, int k0) {
#pragma unroll
        for (int j = 0; j < 4; ++j) {
            int chunk = tid + j * 256;        // 0..1023
            int row   = chunk >> 3;           // 0..127
            int slot  = chunk & 7;            // 8 halves (16B) per slot
            const _Float16* ga =
                X + (size_t)(blockRow + row) * D_EMB + k0 + slot * 8;
            const _Float16* gb =
                X + (size_t)(blockCol + row) * D_EMB + k0 + slot * 8;
            unsigned la = baseA[buf] + (unsigned)(row * RS + slot * 8) * 2u;
            unsigned lb = baseB[buf] + (unsigned)(row * RS + slot * 8) * 2u;
            async_ld16(la, ga);
            async_ld16(lb, gb);
        }
    };

    const int NIT = D_EMB / KS;   // 16 panels
    stage(0, 0);
    wait_async0();
    __syncthreads();

    for (int it = 0; it < NIT; ++it) {
        const int buf = it & 1;
        if (it + 1 < NIT)
            stage(buf ^ 1, (it + 1) * KS);    // async fill of other buffer

#pragma unroll
        for (int sub = 0; sub < 2; ++sub) {   // two 32-K WMMA steps per panel
            v16h a[2];
#pragma unroll
            for (int mi = 0; mi < 2; ++mi) {
                const _Float16* ap =
                    &shA[buf][(waveM * 32 + mi * 16 + nl) * RS + sub * 32 + aKoff];
                union { v16h v; v8h h[2]; } u;
                u.h[0] = *(const v8h*)ap;         // K run 0..7  (+aKoff)
                u.h[1] = *(const v8h*)(ap + 16);  // K run 16..23(+aKoff)
                a[mi] = u.v;
            }
            v16h b[4];
#pragma unroll
            for (int ni = 0; ni < 4; ++ni) {
                const _Float16* bp =
                    &shB[buf][(waveN * 64 + ni * 16 + nl) * RS + sub * 32 + bKoff];
                union { v16h v; v8h h[2]; } u;
                u.h[0] = *(const v8h*)bp;
                u.h[1] = *(const v8h*)(bp + 16);
                b[ni] = u.v;
            }
#pragma unroll
            for (int mi = 0; mi < 2; ++mi)
#pragma unroll
                for (int ni = 0; ni < 4; ++ni)
                    acc[mi][ni] = __builtin_amdgcn_wmma_f32_16x16x32_f16(
                        false, a[mi], false, b[ni],
                        (short)0, acc[mi][ni], false, false);
        }

        wait_async0();     // staged panel landed in LDS
        __syncthreads();   // everyone done reading current buffer
    }

    // ---- fused epilogue: exp, masking, pair sums -------------------------
    const int rowBase = blockRow + waveM * 32;
    const int colBase = blockCol + waveN * 64;

#pragma unroll
    for (int mi = 0; mi < 2; ++mi) {
#pragma unroll
        for (int v = 0; v < 8; ++v) {
            // C/D layout: VGPR v, lanes 0-15 -> M=v, lanes 16-31 -> M=v+8
            const int r = rowBase + mi * 16 + half * 8 + v;
            float part = 0.f;
#pragma unroll
            for (int ni = 0; ni < 4; ++ni) {
                const int c = colBase + ni * 16 + nl;
                float dv = acc[mi][ni][v];
                if (((r & 1) == 0) && (c == r + 1))
                    Dij[r >> 1] = dv;                       // positive-pair sim
                float e = ((c >> 1) == (r >> 1))
                              ? 0.f
                              : __expf(LOSS_MARGIN + dv);   // masked exp
                part += e;
            }
            // reduce across the 16 N-lanes (masks <16 stay within a half)
#pragma unroll
            for (int off = 1; off < 16; off <<= 1)
                part += __shfl_xor(part, off, 32);
            if (nl == 0)
                rowsum[waveM * 32 + mi * 16 + half * 8 + v][waveN] = part;
        }
    }
    __syncthreads();

    if (tid < 64) {   // 64 pairs per 128-row block tile
        float s = rowsum[2 * tid][0] + rowsum[2 * tid][1] +
                  rowsum[2 * tid + 1][0] + rowsum[2 * tid + 1][1];
        atomicAdd(&S[(blockRow >> 1) + tid], s);
    }
}

// ---------------------------------------------------------------------------
// Kernel 3: J = relu(log S - D_ij)^2 ; out = mean(J_tt)/2 + 2*mean(J_st)/2
// ---------------------------------------------------------------------------
__global__ void finalize_kernel(const float* __restrict__ S_tt,
                                const float* __restrict__ D_tt,
                                const float* __restrict__ S_st,
                                const float* __restrict__ D_st,
                                float* __restrict__ out) {
    __shared__ float red[256];
    float att = 0.f, ast = 0.f;
    for (int p = threadIdx.x; p < P_PAIRS; p += 256) {
        float j1 = fmaxf(__logf(S_tt[p]) - D_tt[p], 0.f);
        att += j1 * j1;
        float j2 = fmaxf(__logf(S_st[p]) - D_st[p], 0.f);
        ast += j2 * j2;
    }
    red[threadIdx.x] = att + 2.f * ast;
    __syncthreads();
    for (int s = 128; s > 0; s >>= 1) {
        if (threadIdx.x < s) red[threadIdx.x] += red[threadIdx.x + s];
        __syncthreads();
    }
    if (threadIdx.x == 0)
        out[0] = red[0] / (2.0f * (float)P_PAIRS);
}

// ---------------------------------------------------------------------------
extern "C" void kernel_launch(void* const* d_in, const int* in_sizes, int n_in,
                              void* d_out, int out_size, void* d_ws, size_t ws_size,
                              hipStream_t stream) {
    const float* text  = (const float*)d_in[0];   // [4096,1024] f32
    const float* shape = (const float*)d_in[1];   // [2048,1024] f32

    char* ws = (char*)d_ws;
    const size_t matBytes = (size_t)N_EMB * D_EMB * sizeof(_Float16); // 8 MB
    _Float16* Xtt = (_Float16*)ws;
    _Float16* Xst = (_Float16*)(ws + matBytes);
    float* S_tt = (float*)(ws + 2 * matBytes);
    float* D_tt = S_tt + P_PAIRS;
    float* S_st = D_tt + P_PAIRS;
    float* D_st = S_st + P_PAIRS;

    // zero the S/Dij accumulators (capture-safe async memset)
    hipMemsetAsync(S_tt, 0, 4 * P_PAIRS * sizeof(float), stream);

    build_half_kernel<<<(N_EMB * D_EMB) / 256, 256, 0, stream>>>(
        text, shape, Xtt, Xst);

    dim3 grid(N_EMB / 128, N_EMB / 128);
    gram_loss_kernel<<<grid, 256, 0, stream>>>(Xtt, S_tt, D_tt);
    gram_loss_kernel<<<grid, 256, 0, stream>>>(Xst, S_st, D_st);

    finalize_kernel<<<1, 256, 0, stream>>>(S_tt, D_tt, S_st, D_st,
                                           (float*)d_out);
}
